// RATEncoderLayer_74113955660261
// MI455X (gfx1250) — compile-verified, matrix operand order checked by reading, not run on hardware
//
#include <hip/hip_runtime.h>
#include <hip/hip_bf16.h>

// Problem constants (from reference setup_inputs)
#define B_   8
#define L_   512
#define D_   1024
#define H_   8
#define DH_  128
#define R_   10
#define M_   (B_ * L_)   // 4096 rows
#define DFF_ 4096

typedef __attribute__((ext_vector_type(16))) _Float16 v16h;
typedef __attribute__((ext_vector_type(8)))  float    v8f;
typedef __attribute__((ext_vector_type(4)))  float    f32x4;
typedef __attribute__((ext_vector_type(4)))  _Float16 f16x4;

union FragH { v16h h; f32x4 q[2]; };

// ---------------------------------------------------------------- f32 -> f16
__global__ void k_cvt_f16(const float* __restrict__ s, _Float16* __restrict__ d, int n4) {
    int i = blockIdx.x * blockDim.x + threadIdx.x;
    if (i >= n4) return;
    f32x4 v = ((const f32x4*)s)[i];
    f16x4 o;
    o.x = (_Float16)v.x; o.y = (_Float16)v.y; o.z = (_Float16)v.z; o.w = (_Float16)v.w;
    ((f16x4*)d)[i] = o;
}

// ------------------------------------------------- WMMA GEMM: C = act((A @ W^T + bias) * alpha)
// A: [M,K] f16 row-major.  W: [N,K] f16 row-major (i.e. x @ W^T).
// Block: 256 threads = 8 waves -> 128x128 tile; wave (wm,wn) in 2x4 grid owns 64x32
// (4 M-frags x 2 N-frags of 16x16, K-step 32 per v_wmma_f32_16x16x32_f16).
// Tiles stream in via GLOBAL_LOAD_ASYNC_TO_LDS_B128 (ASYNCcnt) with LDS double buffering;
// the last K-step is peeled so the steady-state loop body is branch-free:
//   issue tile s+1 -> s_wait_asynccnt 4 -> barrier -> ds_load frags -> 8x WMMA -> barrier.
#define BM 128
#define BN 128
#define BK 32
#define LDT 40                    // 32 + 8 halfs pad
#define TILE_H (BM * LDT)         // halfs per tile buffer

__launch_bounds__(256)
__global__ void k_wmma_gemm(const _Float16* __restrict__ A,
                            const _Float16* __restrict__ W,
                            const float* __restrict__ bias,
                            float alpha, int relu,
                            float* __restrict__ C32,
                            _Float16* __restrict__ C16,
                            int Ndim, int Kdim)
{
    __shared__ _Float16 sA[2 * TILE_H];
    __shared__ _Float16 sB[2 * TILE_H];

    const int tid  = threadIdx.x;
    const int lane = tid & 31;
    const int wave = tid >> 5;
    const int wm   = wave >> 2;   // 0..1
    const int wn   = wave & 3;    // 0..3
    const int bm   = blockIdx.y * BM;
    const int bn   = blockIdx.x * BN;

    v8f acc[4][2];
#pragma unroll
    for (int i = 0; i < 4; ++i)
#pragma unroll
        for (int j = 0; j < 2; ++j)
#pragma unroll
            for (int v = 0; v < 8; ++v) acc[i][j][v] = 0.0f;

    // Per-thread async-copy assignment: 2 chunks of 8 halfs (16B) per tile per matrix.
    const int r0  = tid >> 2;           // 0..63
    const int kc0 = (tid & 3) * 8;      // 0,8,16,24 halfs

    // 32-bit LDS byte addresses (low 32 bits of generic LDS pointer = LDS offset)
    const unsigned sAb = (unsigned)(size_t)&sA[0];
    const unsigned sBb = (unsigned)(size_t)&sB[0];

    auto issue_tile = [&](int step, int buf) {
        const int k0 = step * BK;
        const unsigned bufA = sAb + (unsigned)(buf * TILE_H * 2);
        const unsigned bufB = sBb + (unsigned)(buf * TILE_H * 2);
#pragma unroll
        for (int c = 0; c < 2; ++c) {
            const int r = r0 + c * 64;
            const unsigned lofs = (unsigned)((r * LDT + kc0) * 2);
            const unsigned la = bufA + lofs;
            const unsigned lb = bufB + lofs;
            const unsigned long long ga =
                (unsigned long long)(A + (size_t)(bm + r) * Kdim + k0 + kc0);
            const unsigned long long gb =
                (unsigned long long)(W + (size_t)(bn + r) * Kdim + k0 + kc0);
            asm volatile("global_load_async_to_lds_b128 %0, %1, off"
                         :: "v"(la), "v"(ga) : "memory");
            asm volatile("global_load_async_to_lds_b128 %0, %1, off"
                         :: "v"(lb), "v"(gb) : "memory");
        }
    };

    // WMMA A/B per-lane K striping: lanes 0-15 take K [0..7]+[16..23], lanes 16-31 take [8..15]+[24..31]
    const int kofs = (lane & 16) ? 8 : 0;

    auto compute_tile = [&](int buf) {
        const _Float16* tA = sA + buf * TILE_H;
        const _Float16* tB = sB + buf * TILE_H;
        FragH af[4], bf[2];
#pragma unroll
        for (int i = 0; i < 4; ++i) {
            const _Float16* p = tA + (wm * 64 + i * 16 + (lane & 15)) * LDT + kofs;
            af[i].q[0] = *(const f32x4*)p;
            af[i].q[1] = *(const f32x4*)(p + 16);
        }
#pragma unroll
        for (int j = 0; j < 2; ++j) {
            const _Float16* p = tB + (wn * 32 + j * 16 + (lane & 15)) * LDT + kofs;
            bf[j].q[0] = *(const f32x4*)p;
            bf[j].q[1] = *(const f32x4*)(p + 16);
        }
#pragma unroll
        for (int i = 0; i < 4; ++i)
#pragma unroll
            for (int j = 0; j < 2; ++j)
                acc[i][j] = __builtin_amdgcn_wmma_f32_16x16x32_f16(
                    false, af[i].h, false, bf[j].h, (short)0, acc[i][j], false, false);
    };

    const int nsteps = Kdim / BK;   // >= 2 for all our GEMMs

    issue_tile(0, 0);   // prologue: 4 async ops outstanding

    // Steady state: branch-free body, copy of tile s+1 overlaps WMMA on tile s.
    for (int s = 0; s < nsteps - 1; ++s) {
        const int buf = s & 1;
        issue_tile(s + 1, buf ^ 1);                       // 8 outstanding
        // async loads complete in order: <=4 outstanding => tile s has landed
        asm volatile("s_wait_asynccnt 0x4" ::: "memory");
        __syncthreads();   // all waves' tile-s data visible in LDS
        compute_tile(buf);
        __syncthreads();   // all waves done reading buf before it is overwritten
    }

    // Peeled last step.
    asm volatile("s_wait_asynccnt 0x0" ::: "memory");
    __syncthreads();
    compute_tile((nsteps - 1) & 1);

    // Epilogue: VGPR v, lanes 0-15 -> M = v, lanes 16-31 -> M = v + 8; N = lane & 15
    const int rsel = (lane & 16) ? 8 : 0;
    const int col0 = lane & 15;
#pragma unroll
    for (int i = 0; i < 4; ++i)
#pragma unroll
        for (int j = 0; j < 2; ++j) {
            int col  = bn + wn * 32 + j * 16 + col0;
            float bv = bias ? bias[col] : 0.0f;
#pragma unroll
            for (int v = 0; v < 8; ++v) {
                int   row = bm + wm * 64 + i * 16 + v + rsel;
                float x   = (acc[i][j][v] + bv) * alpha;
                if (relu) x = fmaxf(x, 0.0f);
                size_t o = (size_t)row * Ndim + col;
                if (C32) C32[o] = x;
                if (C16) C16[o] = (_Float16)x;
            }
        }
}

// ---------------------------------- tiny projections: out[r,d] = emb[r,:] . W[d,:]   (R*D threads)
__global__ void k_rel_proj(const float* __restrict__ emb, const float* __restrict__ Wt,
                           float* __restrict__ out)
{
    int idx = blockIdx.x * blockDim.x + threadIdx.x;
    if (idx >= R_ * D_) return;
    int r = idx / D_, d = idx - r * D_;
    const f32x4* e = (const f32x4*)(emb + (size_t)r * D_);
    const f32x4* w = (const f32x4*)(Wt + (size_t)d * D_);
    float s = 0.0f;
    for (int k = 0; k < D_ / 4; ++k) {
        f32x4 a = e[k], b = w[k];
        s += a.x * b.x + a.y * b.y + a.z * b.z + a.w * b.w;
    }
    out[idx] = s;
}

// -------------------------------- per-(b,i) histogram of rel_mat row (length L) over R classes
__global__ void k_count(const int* __restrict__ rel, float* __restrict__ cntf)
{
    __shared__ int c[16];
    const int row = blockIdx.x;
    if (threadIdx.x < 16) c[threadIdx.x] = 0;
    __syncthreads();
    const int* p = rel + (size_t)row * L_;
    for (int j = threadIdx.x; j < L_; j += blockDim.x) atomicAdd(&c[p[j] & 15], 1);
    __syncthreads();
    if (threadIdx.x < R_) cntf[(size_t)row * R_ + threadIdx.x] = (float)c[threadIdx.x];
}

// -------------------------------- attention core (const term cancels in softmax):
//   qrel[r] = q_ih . relk_rh ;  pr[r] = cnt[r] e^{qrel[r]} / Z ;  out = xv + pr @ relv
// One block per (b,i), one wave per head; lane owns 4 of dh=128 channels.
__launch_bounds__(256)
__global__ void k_attn(const float* __restrict__ q,   const float* __restrict__ xv,
                       const float* __restrict__ relk, const float* __restrict__ relv,
                       const float* __restrict__ cntf, _Float16* __restrict__ outh)
{
    const int row   = blockIdx.x;
    const int h     = threadIdx.x >> 5;
    const int lane  = threadIdx.x & 31;
    const int dbase = h * DH_ + lane * 4;
    const size_t rb = (size_t)row * D_ + dbase;

    f32x4 q4 = *(const f32x4*)(q + rb);
    float qrel[R_];
#pragma unroll
    for (int r = 0; r < R_; ++r) {
        f32x4 k4 = *(const f32x4*)(relk + (size_t)r * D_ + dbase);
        float s = q4.x * k4.x + q4.y * k4.y + q4.z * k4.z + q4.w * k4.w;
#pragma unroll
        for (int off = 16; off > 0; off >>= 1) s += __shfl_xor(s, off, 32);
        qrel[r] = s;
    }
    float m = qrel[0];
#pragma unroll
    for (int r = 1; r < R_; ++r) m = fmaxf(m, qrel[r]);
    const float* cr = cntf + (size_t)row * R_;
    float w[R_], Z = 0.0f;
#pragma unroll
    for (int r = 0; r < R_; ++r) { w[r] = cr[r] * __expf(qrel[r] - m); Z += w[r]; }
    float inv = 1.0f / Z;

    f32x4 o = *(const f32x4*)(xv + rb);
#pragma unroll
    for (int r = 0; r < R_; ++r) {
        f32x4 v4 = *(const f32x4*)(relv + (size_t)r * D_ + dbase);
        float p = w[r] * inv;
        o.x += p * v4.x; o.y += p * v4.y; o.z += p * v4.z; o.w += p * v4.w;
    }
    f16x4 st;
    st.x = (_Float16)o.x; st.y = (_Float16)o.y; st.z = (_Float16)o.z; st.w = (_Float16)o.w;
    *(f16x4*)(outh + rb) = st;
}

// -------------------------------- fused residual add + LayerNorm (row = D_, 256 thr x 4)
__launch_bounds__(256)
__global__ void k_add_ln(const float* __restrict__ a, const float* __restrict__ b,
                         const float* __restrict__ g, const float* __restrict__ beta,
                         float* __restrict__ out32, _Float16* __restrict__ out16)
{
    __shared__ float sS[8], sQ[8];
    __shared__ float sMean, sRstd;
    const int row  = blockIdx.x;
    const int tid  = threadIdx.x;
    const int lane = tid & 31, wave = tid >> 5;
    const size_t rb = (size_t)row * D_ + tid * 4;

    f32x4 av = *(const f32x4*)(a + rb);
    f32x4 bv = *(const f32x4*)(b + rb);
    f32x4 s4;
    s4.x = av.x + bv.x; s4.y = av.y + bv.y; s4.z = av.z + bv.z; s4.w = av.w + bv.w;

    float s = s4.x + s4.y + s4.z + s4.w;
    float qq = s4.x * s4.x + s4.y * s4.y + s4.z * s4.z + s4.w * s4.w;
#pragma unroll
    for (int off = 16; off > 0; off >>= 1) { s += __shfl_xor(s, off, 32); qq += __shfl_xor(qq, off, 32); }
    if (lane == 0) { sS[wave] = s; sQ[wave] = qq; }
    __syncthreads();
    if (tid == 0) {
        float S = 0.0f, Q = 0.0f;
        for (int i = 0; i < 8; ++i) { S += sS[i]; Q += sQ[i]; }
        float mean = S / (float)D_;
        float var  = Q / (float)D_ - mean * mean;
        sMean = mean;
        sRstd = rsqrtf(var + 1e-5f);
    }
    __syncthreads();
    float mean = sMean, rstd = sRstd;

    int c = tid * 4;
    f32x4 gv = *(const f32x4*)(g + c);
    f32x4 be = *(const f32x4*)(beta + c);
    f32x4 y;
    y.x = (s4.x - mean) * rstd * gv.x + be.x;
    y.y = (s4.y - mean) * rstd * gv.y + be.y;
    y.z = (s4.z - mean) * rstd * gv.z + be.z;
    y.w = (s4.w - mean) * rstd * gv.w + be.w;
    *(f32x4*)(out32 + rb) = y;
    if (out16) {
        f16x4 hh;
        hh.x = (_Float16)y.x; hh.y = (_Float16)y.y; hh.z = (_Float16)y.z; hh.w = (_Float16)y.w;
        *(f16x4*)(out16 + rb) = hh;
    }
}

// ---------------------------------------------------------------------------------------------
extern "C" void kernel_launch(void* const* d_in, const int* in_sizes, int n_in,
                              void* d_out, int out_size, void* d_ws, size_t ws_size,
                              hipStream_t stream)
{
    (void)in_sizes; (void)n_in; (void)out_size; (void)ws_size;

    const float* x   = (const float*)d_in[0];
    const int*   rel = (const int*)  d_in[1];
    const float* Wq  = (const float*)d_in[2];
    const float* bq  = (const float*)d_in[3];
    const float* Wk  = (const float*)d_in[4];
    /* bk = d_in[5] unused: q.xk "const" term cancels in softmax */
    const float* Wv  = (const float*)d_in[6];
    const float* bv  = (const float*)d_in[7];
    const float* Wo  = (const float*)d_in[8];
    const float* bo  = (const float*)d_in[9];
    const float* rke = (const float*)d_in[10];
    const float* rve = (const float*)d_in[11];
    const float* W1  = (const float*)d_in[12];
    const float* b1  = (const float*)d_in[13];
    const float* W2  = (const float*)d_in[14];
    const float* b2  = (const float*)d_in[15];
    const float* g1  = (const float*)d_in[16];
    const float* be1 = (const float*)d_in[17];
    const float* g2  = (const float*)d_in[18];
    const float* be2 = (const float*)d_in[19];
    float* outp = (float*)d_out;

    // Workspace carve-up (all 256B aligned)
    char* p = (char*)d_ws;
    auto take = [&](size_t bytes) -> char* {
        char* r = p;
        p += (bytes + 255) & ~(size_t)255;
        return r;
    };
    _Float16* xh    = (_Float16*)take((size_t)M_ * D_ * 2);
    _Float16* Wqh   = (_Float16*)take((size_t)D_ * D_ * 2);
    _Float16* Wvh   = (_Float16*)take((size_t)D_ * D_ * 2);
    _Float16* Woh   = (_Float16*)take((size_t)D_ * D_ * 2);
    _Float16* W1h   = (_Float16*)take((size_t)DFF_ * D_ * 2);
    _Float16* W2h   = (_Float16*)take((size_t)D_ * DFF_ * 2);
    float*    q32   = (float*)   take((size_t)M_ * D_ * 4);   // reused as o32 after attn
    float*    xv32  = (float*)   take((size_t)M_ * D_ * 4);   // reused as f2 after attn
    float*    relk  = (float*)   take((size_t)R_ * D_ * 4);
    float*    relv  = (float*)   take((size_t)R_ * D_ * 4);
    float*    cntf  = (float*)   take((size_t)M_ * R_ * 4);
    _Float16* outh  = (_Float16*)take((size_t)M_ * D_ * 2);   // reused as x1h after O-proj
    float*    x1_32 = (float*)   take((size_t)M_ * D_ * 4);
    _Float16* hh    = (_Float16*)take((size_t)M_ * DFF_ * 2);
    float*    o32   = q32;
    float*    f2    = xv32;
    _Float16* x1h   = outh;

    dim3 blk(256);
    auto cvt = [&](const float* src, _Float16* dst, size_t n) {
        int n4 = (int)(n / 4);
        k_cvt_f16<<<(n4 + 255) / 256, blk, 0, stream>>>(src, dst, n4);
    };

    // f32 -> f16 for all WMMA operands
    cvt(x,  xh,  (size_t)M_ * D_);
    cvt(Wq, Wqh, (size_t)D_ * D_);
    cvt(Wv, Wvh, (size_t)D_ * D_);
    cvt(Wo, Woh, (size_t)D_ * D_);
    cvt(W1, W1h, (size_t)DFF_ * D_);
    cvt(W2, W2h, (size_t)D_ * DFF_);

    const float qscale = 0.08838834764831845f;  // 1/sqrt(dh), dh=128

    // Q = (x @ Wq^T + bq) * scale ; V = x @ Wv^T + bv
    k_wmma_gemm<<<dim3(D_ / BN, M_ / BM), blk, 0, stream>>>(xh, Wqh, bq, qscale, 0, q32, nullptr, D_, D_);
    k_wmma_gemm<<<dim3(D_ / BN, M_ / BM), blk, 0, stream>>>(xh, Wvh, bv, 1.0f,  0, xv32, nullptr, D_, D_);

    // relk = rel_k_emb @ Wk^T ; relv = rel_v_emb @ Wv^T  (R=10 rows, fp32)
    k_rel_proj<<<(R_ * D_ + 255) / 256, blk, 0, stream>>>(rke, Wk, relk);
    k_rel_proj<<<(R_ * D_ + 255) / 256, blk, 0, stream>>>(rve, Wv, relv);

    // cnt[b,i,r] = |{j : rel_mat[b,i,j] == r}|  (softmax over j collapses to R classes)
    k_count<<<M_, blk, 0, stream>>>(rel, cntf);

    // attention -> f16 context for the O projection
    k_attn<<<M_, blk, 0, stream>>>(q32, xv32, relk, relv, cntf, outh);

    // O projection, then x1 = LN(x + o)
    k_wmma_gemm<<<dim3(D_ / BN, M_ / BM), blk, 0, stream>>>(outh, Woh, bo, 1.0f, 0, o32, nullptr, D_, D_);
    k_add_ln<<<M_, blk, 0, stream>>>(x, o32, g1, be1, x1_32, x1h);

    // FFN: h = relu(x1 @ W1^T + b1) ; f2 = h @ W2^T + b2 ; out = LN(x1 + f2)
    k_wmma_gemm<<<dim3(DFF_ / BN, M_ / BM), blk, 0, stream>>>(x1h, W1h, b1, 1.0f, 1, nullptr, hh, DFF_, D_);
    k_wmma_gemm<<<dim3(D_ / BN, M_ / BM), blk, 0, stream>>>(hh, W2h, b2, 1.0f, 0, f2, nullptr, D_, DFF_);
    k_add_ln<<<M_, blk, 0, stream>>>(x1_32, f2, g2, be2, outp, nullptr);
}
